// TextAutoEncoder_11587821765248
// MI455X (gfx1250) — matrix operand order repaired and checked by reading, not compiled, not run on hardware
//
#include <hip/hip_runtime.h>
#include <cstdint>
#include <cstddef>

// ---------------------------------------------------------------------------
// LSTM autoencoder for MI455X (gfx1250, wave32, WMMA).
// B=64, T=1024, D=256.  Recurrent GEMMs = bf16 WMMA 16x16x32, f32 accum;
// h-state in LDS (bf16), c-state in VGPRs, gates fused on the C-fragment
// layout.  X_t staging uses global_load_async_to_lds_b128 (ASYNCcnt).
// ---------------------------------------------------------------------------

#define DLAT  256
#define B4D   1024   // 4*D  (gate order i,f,c,o)
#define BATCH 64
#define SEQT  1024
#define TBD   (SEQT * BATCH * DLAT)

typedef __attribute__((ext_vector_type(16))) __bf16 v16bf;
typedef __attribute__((ext_vector_type(8)))  __bf16 v8bf;
typedef __attribute__((ext_vector_type(8)))  float  v8f;

enum { MODE_SCALAR = 0, MODE_SEQ = 1, MODE_CONST = 2 };
enum { ACT_RELU = 0, ACT_TANH = 1 };

// float -> bf16 round-to-nearest-even, pure bit ops
__device__ __forceinline__ __bf16 f2bf(float f) {
  unsigned u; __builtin_memcpy(&u, &f, 4);
  unsigned r = (u + 0x7FFFu + ((u >> 16) & 1u)) >> 16;
  unsigned short s = (unsigned short)r;
  __bf16 o; __builtin_memcpy(&o, &s, 2);
  return o;
}

__device__ __forceinline__ float sigmoidf_(float x) {
  return 1.0f / (1.0f + __expf(-x));
}
__device__ __forceinline__ float tanhf_(float x) {
  x = fminf(fmaxf(x, -15.0f), 15.0f);
  float e = __expf(2.0f * x);
  return (e - 1.0f) / (e + 1.0f);
}

// A fragment (16x32 bf16 tile at row m0, k0) in contiguous k order (same as B):
// lane<16: K = k0+0..15; lane>=16: K = k0+16..31; M = lane&15
__device__ __forceinline__ v16bf load_a_frag(const __bf16* base, int m0, int k0, int ld) {
  const int l = (int)(threadIdx.x & 31u);
  const __bf16* p = base + (m0 + (l & 15)) * ld + k0 + ((l >> 4) << 4);
  v8bf lo = *(const v8bf*)(p);
  v8bf hi = *(const v8bf*)(p + 8);
  v16bf a;
#pragma unroll
  for (int e = 0; e < 8; ++e) { a[e] = lo[e]; a[e + 8] = hi[e]; }
  return a;
}

// B fragment (32x16 bf16 tile). Ut pre-packed transposed: Ut[n*DLAT + k].
// lane<16: N = n0+lane, K = k0+0..15; lane>=16: N = n0+lane-16, K = k0+16..31
__device__ __forceinline__ v16bf load_b_frag(const __bf16* Ut, int n0, int k0) {
  const int l = (int)(threadIdx.x & 31u);
  return *(const v16bf*)(Ut + (size_t)(n0 + (l & 15)) * DLAT + k0 + ((l >> 4) << 4));
}

// ---------------------------------------------------------------------------
// Persistent recurrence kernel: one workgroup (32 waves) per direction.
// Wave w owns batch-tiles {w>>4, (w>>4)+2} x hidden-tile (w&15): the two
// patches share all B fragments.  i/f/c/o tiles live in the wave, so the
// LSTM cell update is fully in-register.
// ---------------------------------------------------------------------------
template <int ACT, int MODE>
__global__ __launch_bounds__(1024) void lstm_seq_kernel(
    const __bf16* Ut0, const __bf16* Ut1,                             // recurrent weights (4D x D), per dir (NO restrict: keep B loads in-loop)
    const float* __restrict__ bias0, const float* __restrict__ bias1, // (4D)
    const float* __restrict__ xscalar,                                // MODE_SCALAR: x (B,T)
    const float* __restrict__ w0v0, const float* __restrict__ w0v1,   // MODE_SCALAR: W row (4D)
    const __bf16* __restrict__ Xseq,                                  // MODE_SEQ: (T,B,D) bf16
    const __bf16* Wt,                                                 // MODE_SEQ: packed input weight (4D x D)
    const float* __restrict__ xg0_0, const float* __restrict__ xg0_1, // MODE_CONST: (B,4D)
    float* __restrict__ hout0, float* __restrict__ hout1,             // (T,B,D) f32 or null
    float* __restrict__ hlast,                                        // (B,D) f32 or null
    int T) {
  extern __shared__ char smem[];
  // Mode-specific LDS layout
  constexpr int OFF_X    = 32768;                                  // SEQ only
  constexpr int OFF_BIAS = (MODE == MODE_SEQ) ? 65536 : 32768;
  constexpr int OFF_W0   = OFF_BIAS + 4096;                        // SCALAR
  constexpr int OFF_XS   = OFF_W0 + 4096;                          // SCALAR
  constexpr int OFF_XG   = OFF_BIAS + 4096;                        // CONST (256 KB)

  __bf16* h_bf  = (__bf16*)smem;                 // 32 KB: h state (64x256) bf16
  __bf16* x_bf  = (__bf16*)(smem + OFF_X);       // 32 KB: staged X_t (SEQ)
  float*  biasL = (float*)(smem + OFF_BIAS);     // 4 KB
  float*  w0L   = (float*)(smem + OFF_W0);       // 4 KB (SCALAR)
  float*  xsL   = (float*)(smem + OFF_XS);       // 256 B (SCALAR)
  float*  xg0L  = (float*)(smem + OFF_XG);       // 256 KB (CONST)

  const int dir = (int)blockIdx.x;
  const __bf16* Ut  = dir ? Ut1  : Ut0;
  const float* bias = dir ? bias1 : bias0;
  const float* w0v  = dir ? w0v1  : w0v0;
  const float* xg0  = dir ? xg0_1 : xg0_0;
  float* hout       = dir ? hout1 : hout0;

  const int tid  = (int)threadIdx.x;
  const int wave = tid >> 5;
  const int lane = tid & 31;

  for (int i = tid; i < B4D; i += (int)blockDim.x) {
    biasL[i] = bias[i];
    if (MODE == MODE_SCALAR) w0L[i] = w0v[i];
  }
  if (MODE == MODE_CONST) {  // stage the whole (B,4D) input contribution once
    for (int i = tid; i < BATCH * B4D; i += (int)blockDim.x) xg0L[i] = xg0[i];
  }
  for (int i = tid; i < BATCH * DLAT; i += (int)blockDim.x) h_bf[i] = f2bf(0.0f);
  __syncthreads();

  const int j0  = (wave & 15) << 4;          // hidden tile (shared by both patches)
  const int m0a = ((wave >> 4) + 0) << 4;    // batch tile, patch 0
  const int m0b = ((wave >> 4) + 2) << 4;    // batch tile, patch 1

  v8f cst[2];  // c-state, persistent in VGPRs across all timesteps
#pragma unroll
  for (int pp = 0; pp < 2; ++pp)
#pragma unroll
    for (int r = 0; r < 8; ++r) cst[pp][r] = 0.0f;

  for (int t = 0; t < T; ++t) {
    const int teff = dir ? (T - 1 - t) : t;

    if (MODE == MODE_SCALAR) {
      if (tid < BATCH) xsL[tid] = xscalar[(size_t)tid * T + teff];
    } else if (MODE == MODE_SEQ) {
      // Async DMA of the 32 KB X_t slice into LDS (ASYNCcnt-tracked).
      const unsigned ldsbase = (unsigned)(uintptr_t)(void*)x_bf;
      const unsigned long long gbase =
          (unsigned long long)(uintptr_t)(Xseq + (size_t)teff * BATCH * DLAT);
      const int nvec = BATCH * DLAT * 2 / 16;  // 2048 x 16B
      for (int i = tid; i < nvec; i += (int)blockDim.x) {
        unsigned lds_addr = ldsbase + (unsigned)(i << 4);
        unsigned long long ga = gbase + (unsigned long long)(unsigned)(i << 4);
        asm volatile("global_load_async_to_lds_b128 %0, %1, off"
                     :: "v"(lds_addr), "v"(ga) : "memory");
      }
      if (t + 1 < T) {  // global_prefetch of next step's slice
        const int tn = dir ? (teff - 1) : (teff + 1);
        __builtin_prefetch(Xseq + (size_t)tn * BATCH * DLAT, 0, 1);
      }
      asm volatile("s_wait_asynccnt 0x0" ::: "memory");
    }
    __syncthreads();  // staged data ready; previous-step h_bf writes visible

    v8f acc[2][4];
#pragma unroll
    for (int pp = 0; pp < 2; ++pp)
#pragma unroll
      for (int g = 0; g < 4; ++g) {
        v8f z = {0.f, 0.f, 0.f, 0.f, 0.f, 0.f, 0.f, 0.f};
        acc[pp][g] = z;
      }

    // recurrent contribution: h @ U, K = 256.  K-order rotated by t so the
    // (L2-resident) B-fragment loads cannot be hoisted out of the t-loop.
#pragma unroll
    for (int kk = 0; kk < 8; ++kk) {
      const int k0 = ((kk + t) & 7) << 5;
      v16bf a0 = load_a_frag(h_bf, m0a, k0, DLAT);
      v16bf a1 = load_a_frag(h_bf, m0b, k0, DLAT);
#pragma unroll
      for (int g = 0; g < 4; ++g) {
        v16bf bfrag = load_b_frag(Ut, g * DLAT + j0, k0);
        acc[0][g] = __builtin_amdgcn_wmma_f32_16x16x32_bf16(
            false, a0, false, bfrag, (short)0, acc[0][g], false, false);
        acc[1][g] = __builtin_amdgcn_wmma_f32_16x16x32_bf16(
            false, a1, false, bfrag, (short)0, acc[1][g], false, false);
      }
    }
    if (MODE == MODE_SEQ) {  // input contribution: X_t @ W, K = 256
#pragma unroll
      for (int kk = 0; kk < 8; ++kk) {
        const int k0 = ((kk + t) & 7) << 5;
        v16bf a0 = load_a_frag(x_bf, m0a, k0, DLAT);
        v16bf a1 = load_a_frag(x_bf, m0b, k0, DLAT);
#pragma unroll
        for (int g = 0; g < 4; ++g) {
          v16bf bfrag = load_b_frag(Wt, g * DLAT + j0, k0);
          acc[0][g] = __builtin_amdgcn_wmma_f32_16x16x32_bf16(
              false, a0, false, bfrag, (short)0, acc[0][g], false, false);
          acc[1][g] = __builtin_amdgcn_wmma_f32_16x16x32_bf16(
              false, a1, false, bfrag, (short)0, acc[1][g], false, false);
        }
      }
    }
    __syncthreads();  // all h_bf reads done before epilogue overwrites it

    // fused LSTM cell epilogue on the C-fragment layout:
    // VGPR r, lane l -> M = r + 8*(l>=16), N = l&15
#pragma unroll
    for (int pp = 0; pp < 2; ++pp) {
      const int m0   = pp ? m0b : m0a;
      const int n    = lane & 15;
      const int col  = j0 + n;
      const int rofs = (lane >> 4) << 3;
#pragma unroll
      for (int r = 0; r < 8; ++r) {
        const int b = m0 + r + rofs;
        float vi = acc[pp][0][r] + biasL[col];
        float vf = acc[pp][1][r] + biasL[DLAT + col];
        float vc = acc[pp][2][r] + biasL[2 * DLAT + col];
        float vo = acc[pp][3][r] + biasL[3 * DLAT + col];
        if (MODE == MODE_SCALAR) {
          const float xv = xsL[b];
          vi += xv * w0L[col];
          vf += xv * w0L[DLAT + col];
          vc += xv * w0L[2 * DLAT + col];
          vo += xv * w0L[3 * DLAT + col];
        } else if (MODE == MODE_CONST) {
          const float* xr = xg0L + b * B4D;  // LDS-resident, barrier-fenced
          vi += xr[col];
          vf += xr[DLAT + col];
          vc += xr[2 * DLAT + col];
          vo += xr[3 * DLAT + col];
        }
        const float ig = sigmoidf_(vi);
        const float fg = sigmoidf_(vf);
        const float og = sigmoidf_(vo);
        const float cd = (ACT == ACT_TANH) ? tanhf_(vc) : fmaxf(vc, 0.0f);
        const float c  = fg * cst[pp][r] + ig * cd;
        cst[pp][r] = c;
        const float h = og * ((ACT == ACT_TANH) ? tanhf_(c) : fmaxf(c, 0.0f));
        h_bf[b * DLAT + col] = f2bf(h);
        if (hout) hout[((size_t)teff * BATCH + b) * DLAT + col] = h;
        if (hlast && t == T - 1) hlast[b * DLAT + col] = h;
      }
    }
  }
}

// Transpose+convert weights (K x N f32 row-major) -> (N x K bf16)
__global__ void pack_weight_kernel(const float* __restrict__ U, __bf16* __restrict__ Ut,
                                   int K, int N) {
  int idx = (int)(blockIdx.x * blockDim.x + threadIdx.x);
  if (idx >= K * N) return;
  int n = idx / K;
  int k = idx - n * K;
  Ut[idx] = f2bf(U[(size_t)k * N + n]);
}

__global__ void combine_sum_bf16_kernel(const float* __restrict__ A,
                                        const float* __restrict__ B,
                                        __bf16* __restrict__ out, int n) {
  int i = (int)(blockIdx.x * blockDim.x + threadIdx.x);
  if (i < n) out[i] = f2bf(A[i] + B[i]);
}

__global__ void dec_xg0_kernel(const float* __restrict__ latent,
                               const float* __restrict__ Wf, const float* __restrict__ Wb,
                               float* __restrict__ xg0f, float* __restrict__ xg0b) {
  int g = (int)(blockIdx.x * blockDim.x + threadIdx.x);
  int b = (int)blockIdx.y;
  int dir = (int)blockIdx.z;
  if (g >= B4D) return;
  const float* W = dir ? Wb : Wf;
  float* o = dir ? xg0b : xg0f;
  const float* lr = latent + b * DLAT;
  float s = 0.0f;
  for (int j = 0; j < DLAT; ++j) s += lr[j] * W[(size_t)j * B4D + g];
  o[b * B4D + g] = s;
}

__global__ void dense_out_kernel(const float* __restrict__ A, const float* __restrict__ B,
                                 const float* __restrict__ dW, const float* __restrict__ db,
                                 float* __restrict__ out) {
  int idx = (int)(blockIdx.x * blockDim.x + threadIdx.x);
  if (idx >= SEQT * BATCH) return;
  int t = idx / BATCH;
  int b = idx - t * BATCH;
  const float* pa = A + (size_t)idx * DLAT;
  const float* pb = B + (size_t)idx * DLAT;
  float s = 0.0f;
  for (int j = 0; j < DLAT; ++j) s += (pa[j] + pb[j]) * dW[j];
  out[(size_t)b * SEQT + t] = s + db[0];
}

extern "C" void kernel_launch(void* const* d_in, const int* in_sizes, int n_in,
                              void* d_out, int out_size, void* d_ws, size_t ws_size,
                              hipStream_t stream) {
  (void)in_sizes; (void)n_in; (void)out_size; (void)ws_size;

  const float* x     = (const float*)d_in[0];
  const float* encWf = (const float*)d_in[1];
  const float* encUf = (const float*)d_in[2];
  const float* encbf = (const float*)d_in[3];
  const float* encWb = (const float*)d_in[4];
  const float* encUb = (const float*)d_in[5];
  const float* encbb = (const float*)d_in[6];
  const float* eoW   = (const float*)d_in[7];
  const float* eoU   = (const float*)d_in[8];
  const float* eob   = (const float*)d_in[9];
  const float* decWf = (const float*)d_in[10];
  const float* decUf = (const float*)d_in[11];
  const float* decbf = (const float*)d_in[12];
  const float* decWb = (const float*)d_in[13];
  const float* decUb = (const float*)d_in[14];
  const float* decbb = (const float*)d_in[15];
  const float* dW    = (const float*)d_in[16];
  const float* db    = (const float*)d_in[17];

  char* ws = (char*)d_ws;
  size_t off = 0;
  auto carve = [&](size_t bytes) -> void* {
    void* p = ws + off;
    off += (bytes + 255) & ~(size_t)255;
    return p;
  };
  float*  bufA   = (float*)carve((size_t)TBD * 4);
  float*  bufB   = (float*)carve((size_t)TBD * 4);
  __bf16* Xbf    = (__bf16*)carve((size_t)TBD * 2);
  __bf16* Ut_ef  = (__bf16*)carve((size_t)B4D * DLAT * 2);
  __bf16* Ut_eb  = (__bf16*)carve((size_t)B4D * DLAT * 2);
  __bf16* Wt_eo  = (__bf16*)carve((size_t)B4D * DLAT * 2);
  __bf16* Ut_eo  = (__bf16*)carve((size_t)B4D * DLAT * 2);
  __bf16* Ut_df  = (__bf16*)carve((size_t)B4D * DLAT * 2);
  __bf16* Ut_db  = (__bf16*)carve((size_t)B4D * DLAT * 2);
  float*  xg0f   = (float*)carve((size_t)BATCH * B4D * 4);
  float*  xg0b   = (float*)carve((size_t)BATCH * B4D * 4);
  float*  latent = (float*)carve((size_t)BATCH * DLAT * 4);

  const size_t smem_scalar = 32768 + 4096 + 4096 + 256;   // 41216
  const size_t smem_seq    = 32768 + 32768 + 4096;        // 69632
  const size_t smem_const  = 32768 + 4096 + 262144;       // 299008 (<= 320KB WGP)

  static_assert(32768 + 4096 + 262144 <= 320 * 1024, "LDS over budget");

  // opt into >64KB dynamic LDS (no-op under graph capture; deterministic)
  hipFuncSetAttribute((const void*)&lstm_seq_kernel<ACT_RELU, MODE_SCALAR>,
                      hipFuncAttributeMaxDynamicSharedMemorySize, (int)smem_scalar);
  hipFuncSetAttribute((const void*)&lstm_seq_kernel<ACT_TANH, MODE_SEQ>,
                      hipFuncAttributeMaxDynamicSharedMemorySize, (int)smem_seq);
  hipFuncSetAttribute((const void*)&lstm_seq_kernel<ACT_RELU, MODE_CONST>,
                      hipFuncAttributeMaxDynamicSharedMemorySize, (int)smem_const);

  // 1) pack weights to transposed bf16
  const int pkN = B4D * DLAT;
  dim3 pb(256), pg((pkN + 255) / 256);
  pack_weight_kernel<<<pg, pb, 0, stream>>>(encUf, Ut_ef, DLAT, B4D);
  pack_weight_kernel<<<pg, pb, 0, stream>>>(encUb, Ut_eb, DLAT, B4D);
  pack_weight_kernel<<<pg, pb, 0, stream>>>(eoW,   Wt_eo, DLAT, B4D);
  pack_weight_kernel<<<pg, pb, 0, stream>>>(eoU,   Ut_eo, DLAT, B4D);
  pack_weight_kernel<<<pg, pb, 0, stream>>>(decUf, Ut_df, DLAT, B4D);
  pack_weight_kernel<<<pg, pb, 0, stream>>>(decUb, Ut_db, DLAT, B4D);

  // 2) encoder BiLSTM (relu), scalar input; persistent WGs (fwd,bwd)
  lstm_seq_kernel<ACT_RELU, MODE_SCALAR><<<2, 1024, smem_scalar, stream>>>(
      Ut_ef, Ut_eb, encbf, encbb, x, encWf, encWb,
      nullptr, nullptr, nullptr, nullptr, bufA, bufB, nullptr, SEQT);

  // 3) fwd+bwd sum -> bf16 sequence for the eo LSTM
  combine_sum_bf16_kernel<<<(TBD + 255) / 256, 256, 0, stream>>>(bufA, bufB, Xbf, TBD);

  // 4) eo LSTM (tanh), sequence input, keep only last state -> latent
  lstm_seq_kernel<ACT_TANH, MODE_SEQ><<<1, 1024, smem_seq, stream>>>(
      Ut_eo, Ut_eo, eob, eob, nullptr, nullptr, nullptr, Xbf, Wt_eo,
      nullptr, nullptr, nullptr, nullptr, latent, SEQT);

  // 5) decoder input contribution (constant over t)
  dec_xg0_kernel<<<dim3(4, 64, 2), 256, 0, stream>>>(latent, decWf, decWb, xg0f, xg0b);

  // 6) decoder BiLSTM (relu), constant input
  lstm_seq_kernel<ACT_RELU, MODE_CONST><<<2, 1024, smem_const, stream>>>(
      Ut_df, Ut_db, decbf, decbb, nullptr, nullptr, nullptr, nullptr, nullptr,
      xg0f, xg0b, bufA, bufB, nullptr, SEQT);

  // 7) Dense(1) over fwd+bwd sum -> (B,T) output
  dense_out_kernel<<<(SEQT * BATCH + 255) / 256, 256, 0, stream>>>(
      bufA, bufB, dW, db, (float*)d_out);
}
